// SpikingFeedForward_48335561949682
// MI455X (gfx1250) — matrix operand order repaired and checked by reading, not compiled, not run on hardware
//
#include <hip/hip_runtime.h>

// ---------------------------------------------------------------------------
// SpikingFeedForward for MI455X (gfx1250, wave32, WMMA + async LDS staging)
//   GEMM1(f16-split, 3x wmma) + PLIF scan fused  ->  s1 (f16, exact 0/1)
//   GEMM2(s1 exact f16, W2 f16-split, 2x wmma) + PLIF scan fused -> out f32
//   A-operand staged global->LDS with GLOBAL_LOAD_ASYNC_TO_LDS_B128,
//   double buffered, shared by the 4 waves of a block (4x less A traffic).
// ---------------------------------------------------------------------------

typedef __attribute__((ext_vector_type(16))) _Float16 v16h;
typedef __attribute__((ext_vector_type(8)))  float    v8f;

#define T_STEPS 4
#define BN      8192          // B*N rows per timestep
#define TBN     32768         // T*B*N
#define DDIM    512
#define HDIM    2048
#define NTILE   2             // column tiles per wave
#define WAVES   4             // waves per block ( == T_STEPS: wave w stages t=w )

// ---- CDNA5 async global->LDS copy (16B per lane, tracked by ASYNCcnt) -----
__device__ __forceinline__ void async_b128(void* lds_dst, const void* gsrc) {
  unsigned lds_off = (unsigned)(uintptr_t)lds_dst;   // flat LDS addr[31:0] == LDS offset
  asm volatile("global_load_async_to_lds_b128 %0, %1, off"
               :: "v"(lds_off), "v"(gsrc) : "memory");
}

__device__ __forceinline__ void wait_async0() {
#if __has_builtin(__builtin_amdgcn_s_wait_asynccnt)
  __builtin_amdgcn_s_wait_asynccnt(0);
#else
  asm volatile("s_wait_asynccnt 0x0" ::: "memory");
#endif
}

__device__ __forceinline__ v8f wmma_f16(v16h a, v16h b, v8f c) {
  return __builtin_amdgcn_wmma_f32_16x16x32_f16(false, a, false, b,
                                                (short)0, c, false, false);
}

__device__ __forceinline__ v8f vzero8() {
  v8f z;
#pragma unroll
  for (int i = 0; i < 8; ++i) z[i] = 0.0f;
  return z;
}

// f32 -> (hi,lo) f16 split:  v ~= (float)hi + (float)lo, rel err ~ eps_f16^2
__global__ void split_f32_to_f16_hilo(const float* __restrict__ src,
                                      _Float16* __restrict__ hi,
                                      _Float16* __restrict__ lo, long n) {
  long i = (long)blockIdx.x * blockDim.x + threadIdx.x;
  long stride = (long)gridDim.x * blockDim.x;
  for (; i < n; i += stride) {
    float v = src[i];
    _Float16 h = (_Float16)v;
    hi[i] = h;
    lo[i] = (_Float16)(v - (float)h);
  }
}

// ---------------------------------------------------------------------------
// Kernel 1: h = x @ W1^T (split-f16, 3 wmma/K-step), PLIF over T fused.
// LDS layout: At[buf][t][hi|lo][512 halves]; lane L's 32B fragment at half
// offset L*16 (chunks L*16 and L*16+8), i.e. already in WMMA A-fragment order.
// ---------------------------------------------------------------------------
__global__ void __launch_bounds__(WAVES * 32)
ffn1_gemm_plif(const _Float16* __restrict__ xh, const _Float16* __restrict__ xl,
               const _Float16* __restrict__ w1h, const _Float16* __restrict__ w1l,
               const float* __restrict__ alpha, _Float16* __restrict__ s1) {
  __shared__ __align__(32) _Float16 At[2][T_STEPS][2][512];   // 16 KB

  const int lane = threadIdx.x & 31;
  const int wave = threadIdx.x >> 5;
  const int m0 = blockIdx.x * 16;                               // row tile in [0,BN)
  const int n0 = (blockIdx.y * WAVES + wave) * (16 * NTILE);    // col tile in [0,HDIM)

  const int lm = lane & 15;
  const int hg = lane >> 4;        // lane group (0 or 1)
  const int ak = hg << 4;          // A K-offset: 0 or 16
  const int bk = hg << 4;          // B K-offset: 0 or 16

  // loader: wave w stages timestep t==w; addresses match the A-fragment map
  const _Float16* ga_h = xh + ((size_t)wave * BN + m0 + lm) * DDIM + ak;
  const _Float16* ga_l = xl + ((size_t)wave * BN + m0 + lm) * DDIM + ak;

  const _Float16* pbh[NTILE];
  const _Float16* pbl[NTILE];
#pragma unroll
  for (int j = 0; j < NTILE; ++j) {
    pbh[j] = w1h + (size_t)(n0 + j * 16 + lm) * DDIM + bk;  // W1 row-major == B^T
    pbl[j] = w1l + (size_t)(n0 + j * 16 + lm) * DDIM + bk;
  }

  v8f acc[T_STEPS][NTILE];
#pragma unroll
  for (int t = 0; t < T_STEPS; ++t)
#pragma unroll
    for (int j = 0; j < NTILE; ++j) acc[t][j] = vzero8();

  auto stage = [&](int buf, int k0) {
    async_b128(&At[buf][wave][0][lane * 16],     ga_h + k0);
    async_b128(&At[buf][wave][0][lane * 16 + 8], ga_h + k0 + 8);
    async_b128(&At[buf][wave][1][lane * 16],     ga_l + k0);
    async_b128(&At[buf][wave][1][lane * 16 + 8], ga_l + k0 + 8);
  };

  stage(0, 0);
  int p = 0;
  for (int i = 0; i < DDIM / 32; ++i, p ^= 1) {
    const int k0 = i * 32;
    wait_async0();          // my 4 copies for buf p have landed
    __syncthreads();        // everyone's copies visible; prev reads of buf p^1 done
    if (i + 1 < DDIM / 32) stage(p ^ 1, k0 + 32);

    v16h bh[NTILE], bl[NTILE];
#pragma unroll
    for (int j = 0; j < NTILE; ++j) {
      bh[j] = *(const v16h*)(pbh[j] + k0);   // 32B contiguous global
      bl[j] = *(const v16h*)(pbl[j] + k0);
    }
#pragma unroll
    for (int t = 0; t < T_STEPS; ++t) {
      v16h ah = *(const v16h*)&At[p][t][0][lane * 16];   // 32B contiguous LDS
      v16h al = *(const v16h*)&At[p][t][1][lane * 16];
#pragma unroll
      for (int j = 0; j < NTILE; ++j) {
        acc[t][j] = wmma_f16(ah, bh[j], acc[t][j]);   // hi*hi
        acc[t][j] = wmma_f16(ah, bl[j], acc[t][j]);   // hi*lo
        acc[t][j] = wmma_f16(al, bh[j], acc[t][j]);   // lo*hi
      }
    }
  }

  // PLIF scan over T in registers; hard reset; write spikes (exact in f16).
  const float a = alpha[0];
  float v[NTILE][8];
#pragma unroll
  for (int j = 0; j < NTILE; ++j)
#pragma unroll
    for (int r = 0; r < 8; ++r) v[j][r] = 0.0f;

#pragma unroll
  for (int t = 0; t < T_STEPS; ++t) {
#pragma unroll
    for (int j = 0; j < NTILE; ++j) {
#pragma unroll
      for (int r = 0; r < 8; ++r) {
        float vv = v[j][r];
        vv += a * (acc[t][j][r] - vv);
        float s = (vv >= 1.0f) ? 1.0f : 0.0f;   // heaviside(v - v_th), v_th=1
        v[j][r] = (s != 0.0f) ? 0.0f : vv;      // hard reset
        const size_t row = (size_t)(t * BN + m0 + (hg << 3) + r);
        s1[row * HDIM + (size_t)(n0 + j * 16 + lm)] = (_Float16)s;
      }
    }
  }
}

// ---------------------------------------------------------------------------
// Kernel 2: y = s1 @ W2^T (s1 exact f16, W2 split -> 2 wmma/K-step),
//           PLIF over T fused, spikes written as f32 to d_out.
// ---------------------------------------------------------------------------
__global__ void __launch_bounds__(WAVES * 32)
ffn2_gemm_plif(const _Float16* __restrict__ s1,
               const _Float16* __restrict__ w2h, const _Float16* __restrict__ w2l,
               const float* __restrict__ alpha, float* __restrict__ out) {
  __shared__ __align__(32) _Float16 At[2][T_STEPS][512];   // 8 KB

  const int lane = threadIdx.x & 31;
  const int wave = threadIdx.x >> 5;
  const int m0 = blockIdx.x * 16;                               // row tile in [0,BN)
  const int n0 = (blockIdx.y * WAVES + wave) * (16 * NTILE);    // col tile in [0,DDIM)

  const int lm = lane & 15;
  const int hg = lane >> 4;
  const int ak = hg << 4;
  const int bk = hg << 4;

  const _Float16* ga = s1 + ((size_t)wave * BN + m0 + lm) * HDIM + ak;

  const _Float16* pbh[NTILE];
  const _Float16* pbl[NTILE];
#pragma unroll
  for (int j = 0; j < NTILE; ++j) {
    pbh[j] = w2h + (size_t)(n0 + j * 16 + lm) * HDIM + bk;  // W2 row-major == B^T
    pbl[j] = w2l + (size_t)(n0 + j * 16 + lm) * HDIM + bk;
  }

  v8f acc[T_STEPS][NTILE];
#pragma unroll
  for (int t = 0; t < T_STEPS; ++t)
#pragma unroll
    for (int j = 0; j < NTILE; ++j) acc[t][j] = vzero8();

  auto stage = [&](int buf, int k0) {
    async_b128(&At[buf][wave][lane * 16],     ga + k0);
    async_b128(&At[buf][wave][lane * 16 + 8], ga + k0 + 8);
  };

  stage(0, 0);
  int p = 0;
  for (int i = 0; i < HDIM / 32; ++i, p ^= 1) {
    const int k0 = i * 32;
    wait_async0();
    __syncthreads();
    if (i + 1 < HDIM / 32) stage(p ^ 1, k0 + 32);

    v16h bh[NTILE], bl[NTILE];
#pragma unroll
    for (int j = 0; j < NTILE; ++j) {
      bh[j] = *(const v16h*)(pbh[j] + k0);
      bl[j] = *(const v16h*)(pbl[j] + k0);
    }
#pragma unroll
    for (int t = 0; t < T_STEPS; ++t) {
      v16h av = *(const v16h*)&At[p][t][lane * 16];
#pragma unroll
      for (int j = 0; j < NTILE; ++j) {
        acc[t][j] = wmma_f16(av, bh[j], acc[t][j]);   // s1 * W2_hi
        acc[t][j] = wmma_f16(av, bl[j], acc[t][j]);   // s1 * W2_lo
      }
    }
  }

  const float a = alpha[0];
  float v[NTILE][8];
#pragma unroll
  for (int j = 0; j < NTILE; ++j)
#pragma unroll
    for (int r = 0; r < 8; ++r) v[j][r] = 0.0f;

#pragma unroll
  for (int t = 0; t < T_STEPS; ++t) {
#pragma unroll
    for (int j = 0; j < NTILE; ++j) {
#pragma unroll
      for (int r = 0; r < 8; ++r) {
        float vv = v[j][r];
        vv += a * (acc[t][j][r] - vv);
        float s = (vv >= 1.0f) ? 1.0f : 0.0f;
        v[j][r] = (s != 0.0f) ? 0.0f : vv;
        const size_t row = (size_t)(t * BN + m0 + (hg << 3) + r);
        out[row * DDIM + (size_t)(n0 + j * 16 + lm)] = s;
      }
    }
  }
}

// ---------------------------------------------------------------------------
extern "C" void kernel_launch(void* const* d_in, const int* in_sizes, int n_in,
                              void* d_out, int out_size, void* d_ws, size_t ws_size,
                              hipStream_t stream) {
  (void)in_sizes; (void)n_in; (void)out_size;
  const float* x  = (const float*)d_in[0];   // [T,B,N,D] f32
  const float* W1 = (const float*)d_in[1];   // [H,D]     f32
  const float* W2 = (const float*)d_in[2];   // [D,H]     f32
  const float* a1 = (const float*)d_in[3];   // [1]
  const float* a2 = (const float*)d_in[4];   // [1]
  float* out = (float*)d_out;                // [T,B,N,D] f32

  // Workspace layout (~210 MB total, all chunks 32B+ aligned)
  const size_t XN = (size_t)TBN * DDIM;      // 16,777,216
  const size_t WN = (size_t)HDIM * DDIM;     // 1,048,576
  const size_t SN = (size_t)TBN * HDIM;      // 67,108,864
  char* ws = (char*)d_ws;
  _Float16* xh  = (_Float16*)ws;               ws += XN * sizeof(_Float16);
  _Float16* xl  = (_Float16*)ws;               ws += XN * sizeof(_Float16);
  _Float16* w1h = (_Float16*)ws;               ws += WN * sizeof(_Float16);
  _Float16* w1l = (_Float16*)ws;               ws += WN * sizeof(_Float16);
  _Float16* w2h = (_Float16*)ws;               ws += WN * sizeof(_Float16);
  _Float16* w2l = (_Float16*)ws;               ws += WN * sizeof(_Float16);
  _Float16* s1  = (_Float16*)ws;               ws += SN * sizeof(_Float16);
  (void)ws_size;  // requires ~210 MB

  // Precision prep: split f32 operands into f16 hi/lo pairs.
  split_f32_to_f16_hilo<<<8192, 256, 0, stream>>>(x,  xh,  xl,  (long)XN);
  split_f32_to_f16_hilo<<<2048, 256, 0, stream>>>(W1, w1h, w1l, (long)WN);
  split_f32_to_f16_hilo<<<2048, 256, 0, stream>>>(W2, w2h, w2l, (long)WN);

  // GEMM1 + PLIF1: grid = (row tiles, col supertiles)
  dim3 g1(BN / 16, HDIM / (16 * NTILE * WAVES));   // (512, 16)
  ffn1_gemm_plif<<<g1, WAVES * 32, 0, stream>>>(xh, xl, w1h, w1l, a1, s1);

  // GEMM2 + PLIF2
  dim3 g2(BN / 16, DDIM / (16 * NTILE * WAVES));   // (512, 4)
  ffn2_gemm_plif<<<g2, WAVES * 32, 0, stream>>>(s1, w2h, w2l, a2, out);
}
